// Model4BD_ch_89584427860261
// MI455X (gfx1250) — matrix operand order repaired and checked by reading, not compiled, not run on hardware
//
#include <hip/hip_runtime.h>
#include <hip/hip_bf16.h>

#define NEXP  210
#define NSML  7
#define NET   217
#define BATCH 8192
#define H     128
#define BM    128
#define KH    136   // padded K-stride (halves) for 128-K layers: 272B, 16B-aligned, conflict-free
#define K1    48    // padded K-stride (halves) for layer-1 (K=32 logical): 96B, 16B-aligned

typedef __attribute__((ext_vector_type(16))) _Float16 v16h;
typedef __attribute__((ext_vector_type(8)))  _Float16 v8h;
typedef __attribute__((ext_vector_type(8)))  float    v8f;

__device__ __forceinline__ float tanx(float x) {
#if __has_builtin(__builtin_amdgcn_tanhf)
  return __builtin_amdgcn_tanhf(x);
#else
  return tanhf(x);
#endif
}

// ---- CDNA5 async global->LDS staging (ASYNCcnt-tracked, no VGPR round trip) ----
__device__ __forceinline__ void wait_async0() {
#if __has_builtin(__builtin_amdgcn_s_wait_asynccnt)
  __builtin_amdgcn_s_wait_asynccnt(0);
#else
  asm volatile("s_wait_asynccnt 0x0" ::: "memory");
#endif
}

__device__ __forceinline__ void async_b128(unsigned lds_addr, unsigned voff,
                                           unsigned long long sbase) {
  // GVS mode: mem = SADDR(base) + VADDR(32-bit offset); vdst = LDS byte address
  asm volatile("global_load_async_to_lds_b128 %0, %1, %2 offset:0"
               :: "v"(lds_addr), "v"(voff), "s"(sbase)
               : "memory");
}

// copy n16*16 bytes from gsrc into LDS at lds_dst, 16B per lane per issue
__device__ __forceinline__ void stage_async(void* lds_dst, const void* gsrc,
                                            int n16, int tid) {
  const unsigned lbase = (unsigned)(unsigned long long)(uintptr_t)lds_dst;
  const unsigned long long g = (unsigned long long)(uintptr_t)gsrc;
  for (int i = tid; i < n16; i += 256)
    async_b128(lbase + (unsigned)i * 16u, (unsigned)i * 16u, g);
}

// ---- WMMA fragment loads per CDNA5 ISA 7.12.2 layouts (wave32) ----
// A 16x32 f16: lane<16 holds row=lane, K = kb+[0..7] (v0..3) and kb+[16..23] (v4..7);
//              lane>=16 holds row=lane-16, K = kb+[8..15] and kb+[24..31].
__device__ __forceinline__ v16h ldA(const _Float16* h, int row, int kb, int lane) {
  const _Float16* p = h + row * KH + kb + ((lane >> 4) << 3);
  v16h a;
  ((v8h*)&a)[0] = *(const v8h*)(p);
  ((v8h*)&a)[1] = *(const v8h*)(p + 16);
  return a;
}
// B 32x16 f16 (w stored N-major [n][k]): lane<16: col=lane, K=kb+[0..7] and kb+[16..23];
//                                        lane>=16: col=lane-16, K=kb+[8..15] and kb+[24..31].
__device__ __forceinline__ v16h ldB(const _Float16* w, int col, int kstride, int kb, int lane) {
  const _Float16* p = w + col * kstride + kb + ((lane >> 4) << 3);
  v16h b;
  ((v8h*)&b)[0] = *(const v8h*)(p);
  ((v8h*)&b)[1] = *(const v8h*)(p + 16);
  return b;
}

// ---------------- Kernel A: weight convert + swizzle (f32 -> f16 LDS images) -------------
__global__ __launch_bounds__(256) void cvt_weights_k(
    const float* __restrict__ W1,  const float* __restrict__ Wh,
    const float* __restrict__ Wo,  const float* __restrict__ b1,
    const float* __restrict__ bh,  const float* __restrict__ bo,
    const float* __restrict__ W1s, const float* __restrict__ Whs,
    const float* __restrict__ Wos, const float* __restrict__ b1s,
    const float* __restrict__ bhs, const float* __restrict__ bos,
    _Float16* __restrict__ wsL1, _Float16* __restrict__ wsL2,
    _Float16* __restrict__ wsL3, float* __restrict__ wsTail)
{
  const int e = blockIdx.x, region = blockIdx.y, tid = threadIdx.x;
  const bool sm = (e >= NEXP);
  const int  s  = e - NEXP;
  if (region == 0) {
    _Float16* img = wsL1 + (size_t)e * H * K1;                // [n][k], k<kv valid
    const int kv = sm ? 2 : 8;
    const float* src = sm ? (W1s + (size_t)s * 2 * H) : (W1 + (size_t)e * 8 * H); // [k][n]
    for (int i = tid; i < H * K1; i += 256) {
      int n = i / K1, k = i % K1;
      img[i] = (_Float16)((k < kv) ? src[k * H + n] : 0.f);
    }
  } else if (region <= 2) {
    const int l = region - 1;
    _Float16* img = (l == 0 ? wsL2 : wsL3) + (size_t)e * H * KH;
    const float* src = sm ? (Whs + ((size_t)(l * NSML + s) * H) * H)
                          : (Wh  + ((size_t)(l * NEXP + e) * H) * H); // [k][n]
    for (int i = tid; i < H * KH; i += 256) {
      int n = i / KH, k = i % KH;
      img[i] = (_Float16)((k < H) ? src[k * H + n] : 0.f);
    }
  } else {
    float* t = wsTail + (size_t)e * 520; // [0:128)=b1 [128:256)=bh0 [256:384)=bh1 [384:512)=Wo [512]=bo
    for (int i = tid; i < 513; i += 256) {
      float v;
      if      (i < 128) v = sm ? b1s[s * H + i]                    : b1[e * H + i];
      else if (i < 256) v = sm ? bhs[(0 * NSML + s) * H + (i-128)] : bh[(0 * NEXP + e) * H + (i-128)];
      else if (i < 384) v = sm ? bhs[(1 * NSML + s) * H + (i-256)] : bh[(1 * NEXP + e) * H + (i-256)];
      else if (i < 512) v = sm ? Wos[s * H + (i-384)]              : Wo[e * H + (i-384)];
      else              v = sm ? bos[s]                            : bo[e];
      t[i] = v;
    }
  }
}

// ---------------- Kernel B: fused 3-layer expert MLP via WMMA ----------------------------
__global__ __launch_bounds__(256) void expert_fwd_k(
    const float* __restrict__ F_dist, const float* __restrict__ F_cos,
    const float* __restrict__ F_sin,
    const float* __restrict__ Zd, const float* __restrict__ Zc, const float* __restrict__ Zs,
    const int* __restrict__ I34, const int* __restrict__ I5,
    const _Float16* __restrict__ wsL1, const _Float16* __restrict__ wsL2,
    const _Float16* __restrict__ wsL3, const float* __restrict__ wsTail,
    float* __restrict__ partial)
{
  __shared__ _Float16 hA[BM * KH];   // 34816 B
  __shared__ _Float16 hB[BM * KH];   // 34816 B
  __shared__ _Float16 WA[H * KH];    // 34816 B (layer1 uses first H*K1, layer3 full)
  __shared__ _Float16 WB[H * KH];    // 34816 B (layer2)
  __shared__ float    tl[516];       // biases + Wo + bo

  const int e = blockIdx.y, tile = blockIdx.x, tid = threadIdx.x;
  const int lane = tid & 31, wave = tid >> 5;
  const int rowbase = tile * BM;
  const int mbase = wave * 16;
  const int nlo = lane & 15, hiv = lane >> 4;

  // ---- phase 0: issue L1-weight asyncs, stage tail, build F tile ----
  stage_async(WA, wsL1 + (size_t)e * H * K1, (H * K1) / 8, tid);   // 768 x 16B
  { const float* ts = wsTail + (size_t)e * 520;
    for (int i = tid; i < 513; i += 256) tl[i] = ts[i]; }

  // normalized, gathered F tile (BM x 32, zero-padded K) into hA
  if (tid < BM) {
    const int row = rowbase + tid;
    _Float16* fr = hA + tid * KH;
    float v[8]; int nv;
    if (e < NEXP) {
      #pragma unroll
      for (int j = 0; j < 6; ++j) {
        int ix = I34[e * 6 + j];
        v[j] = (F_dist[(size_t)row * 45 + ix] - Zd[ix]) / Zd[45 + ix];
      }
      v[6] = (F_cos[(size_t)row * NEXP + e] - Zc[e]) / Zc[NEXP + e];
      v[7] = (F_sin[(size_t)row * NEXP + e] - Zs[e]) / Zs[NEXP + e];
      nv = 8;
    } else {
      int i5 = I5[e - NEXP];
      v[0] = (F_cos[(size_t)row * NEXP + i5] - Zc[i5]) / Zc[NEXP + i5];
      v[1] = (F_sin[(size_t)row * NEXP + i5] - Zs[i5]) / Zs[NEXP + i5];
      nv = 2;
    }
    #pragma unroll
    for (int k = 0; k < 32; ++k) fr[k] = (k < nv) ? (_Float16)v[k] : (_Float16)0.f;
  }
  wait_async0();
  __syncthreads();

  v8f acc[8];

  // ---- phase 1: issue L2-weight asyncs, compute Layer 1 (F x W1 + b1, tanh -> hB) ----
  stage_async(WB, wsL2 + (size_t)e * H * KH, (H * KH) / 8, tid);   // 2176 x 16B
  #pragma unroll
  for (int nt = 0; nt < 8; ++nt) { float b = tl[nt * 16 + nlo];
    #pragma unroll
    for (int q = 0; q < 8; ++q) acc[nt][q] = b; }
  {
    v16h a = ldA(hA, mbase + nlo, 0, lane);
    #pragma unroll
    for (int nt = 0; nt < 8; ++nt) {
      v16h b = ldB(WA, nt * 16 + nlo, K1, 0, lane);
      acc[nt] = __builtin_amdgcn_wmma_f32_16x16x32_f16(false, a, false, b, (short)0, acc[nt], false, false);
    }
  }
  #pragma unroll
  for (int nt = 0; nt < 8; ++nt)
    #pragma unroll
    for (int q = 0; q < 8; ++q)
      hB[(mbase + q + hiv * 8) * KH + nt * 16 + nlo] = (_Float16)tanx(acc[nt][q]);
  wait_async0();
  __syncthreads();

  // ---- phase 2: issue L3-weight asyncs into WA, compute Layer 2 (hB x Wh0 -> hA) ----
  stage_async(WA, wsL3 + (size_t)e * H * KH, (H * KH) / 8, tid);
  #pragma unroll
  for (int nt = 0; nt < 8; ++nt) { float b = tl[128 + nt * 16 + nlo];
    #pragma unroll
    for (int q = 0; q < 8; ++q) acc[nt][q] = b; }
  #pragma unroll
  for (int ks = 0; ks < 4; ++ks) {
    v16h a = ldA(hB, mbase + nlo, ks * 32, lane);
    #pragma unroll
    for (int nt = 0; nt < 8; ++nt) {
      v16h b = ldB(WB, nt * 16 + nlo, KH, ks * 32, lane);
      acc[nt] = __builtin_amdgcn_wmma_f32_16x16x32_f16(false, a, false, b, (short)0, acc[nt], false, false);
    }
  }
  #pragma unroll
  for (int nt = 0; nt < 8; ++nt)
    #pragma unroll
    for (int q = 0; q < 8; ++q)
      hA[(mbase + q + hiv * 8) * KH + nt * 16 + nlo] = (_Float16)tanx(acc[nt][q]);
  wait_async0();
  __syncthreads();

  // ---- phase 3: Layer 3 (hA x Wh1 + bh1, tanh) fused with output dot (Wo, bo) ----
  #pragma unroll
  for (int nt = 0; nt < 8; ++nt) { float b = tl[256 + nt * 16 + nlo];
    #pragma unroll
    for (int q = 0; q < 8; ++q) acc[nt][q] = b; }
  #pragma unroll
  for (int ks = 0; ks < 4; ++ks) {
    v16h a = ldA(hA, mbase + nlo, ks * 32, lane);
    #pragma unroll
    for (int nt = 0; nt < 8; ++nt) {
      v16h b = ldB(WA, nt * 16 + nlo, KH, ks * 32, lane);
      acc[nt] = __builtin_amdgcn_wmma_f32_16x16x32_f16(false, a, false, b, (short)0, acc[nt], false, false);
    }
  }
  float rs[8];
  #pragma unroll
  for (int q = 0; q < 8; ++q) rs[q] = 0.f;
  #pragma unroll
  for (int nt = 0; nt < 8; ++nt) {
    float wo = tl[384 + nt * 16 + nlo];
    #pragma unroll
    for (int q = 0; q < 8; ++q) rs[q] += tanx(acc[nt][q]) * wo;
  }
  // reduce the 16 N-lanes of each half-wave (rows stay within 16-lane groups)
  #pragma unroll
  for (int q = 0; q < 8; ++q) {
    rs[q] += __shfl_xor(rs[q], 1, 32);
    rs[q] += __shfl_xor(rs[q], 2, 32);
    rs[q] += __shfl_xor(rs[q], 4, 32);
    rs[q] += __shfl_xor(rs[q], 8, 32);
  }
  if (nlo == 0) {
    float bov = tl[512];
    #pragma unroll
    for (int q = 0; q < 8; ++q)
      partial[(size_t)e * BATCH + rowbase + mbase + q + hiv * 8] = rs[q] + bov;
  }
}

// ---------------- Kernel C: sum partials over experts ------------------------------------
__global__ __launch_bounds__(256) void reduce_k(const float* __restrict__ partial,
                                                float* __restrict__ out)
{
  const int r = blockIdx.x * 256 + threadIdx.x;
  float s = 0.f;
  for (int e = 0; e < NET; ++e) s += partial[(size_t)e * BATCH + r];
  out[r] = s;
}

extern "C" void kernel_launch(void* const* d_in, const int* in_sizes, int n_in,
                              void* d_out, int out_size, void* d_ws, size_t ws_size,
                              hipStream_t stream)
{
  const float* F_dist = (const float*)d_in[0];
  const float* F_cos  = (const float*)d_in[1];
  const float* F_sin  = (const float*)d_in[2];
  // d_in[3] F_angle: unused by the reference computation
  const float* Zd  = (const float*)d_in[4];
  const float* Zc  = (const float*)d_in[5];
  const float* Zs  = (const float*)d_in[6];
  const int*   I34 = (const int*)d_in[7];
  const int*   I5  = (const int*)d_in[8];
  const float* W1  = (const float*)d_in[9];
  const float* b1  = (const float*)d_in[10];
  const float* Wh  = (const float*)d_in[11];
  const float* bh  = (const float*)d_in[12];
  const float* Wo  = (const float*)d_in[13];
  const float* bo  = (const float*)d_in[14];
  const float* W1s = (const float*)d_in[15];
  const float* b1s = (const float*)d_in[16];
  const float* Whs = (const float*)d_in[17];
  const float* bhs = (const float*)d_in[18];
  const float* Wos = (const float*)d_in[19];
  const float* bos = (const float*)d_in[20];

  char* ws = (char*)d_ws;
  const size_t SZ_L1 = (size_t)NET * H * K1 * sizeof(_Float16);   // 2,666,496
  const size_t SZ_LH = (size_t)NET * H * KH * sizeof(_Float16);   // 7,555,072
  const size_t SZ_TL = (size_t)NET * 520 * sizeof(float);         //   451,360
  _Float16* wsL1   = (_Float16*)(ws);
  _Float16* wsL2   = (_Float16*)(ws + SZ_L1);
  _Float16* wsL3   = (_Float16*)(ws + SZ_L1 + SZ_LH);
  float*    wsTail = (float*)   (ws + SZ_L1 + 2 * SZ_LH);
  float*    partial= (float*)   (ws + SZ_L1 + 2 * SZ_LH + SZ_TL); // 217*8192 f32

  cvt_weights_k<<<dim3(NET, 4), 256, 0, stream>>>(
      W1, Wh, Wo, b1, bh, bo, W1s, Whs, Wos, b1s, bhs, bos,
      wsL1, wsL2, wsL3, wsTail);

  expert_fwd_k<<<dim3(BATCH / BM, NET), 256, 0, stream>>>(
      F_dist, F_cos, F_sin, Zd, Zc, Zs, I34, I5,
      wsL1, wsL2, wsL3, wsTail, partial);

  reduce_k<<<dim3(BATCH / 256), 256, 0, stream>>>(partial, (float*)d_out);
}
